// FP8Linear_28372553957505
// MI455X (gfx1250) — 8-bit matrix operand order repaired and checked by reading, compiled, not run on hardware
//
#include <hip/hip_runtime.h>
#include <hip/hip_bf16.h>
#include <stdint.h>

typedef int   v8i_t __attribute__((ext_vector_type(8)));
typedef float v8f_t __attribute__((ext_vector_type(8)));

#define KDIM 4096
#define NDIM 4096
#define KT   (KDIM / 64)
#define NB64 (NDIM / 64)

static __device__ __forceinline__ unsigned f32_to_e4m3_sw(float f) {
    unsigned u = __float_as_uint(f);
    unsigned sgn = (u >> 31) << 7;
    float a = fabsf(f);
    if (!(a == a)) return sgn | 0x7Fu;
    if (a >= 448.0f) return sgn | 0x7Eu;
    int ex = (int)((__float_as_uint(a) >> 23) & 0xFF) - 127;
    int ulp_e = ex - 3; if (ulp_e < -9) ulp_e = -9;
    float ulp = ldexpf(1.0f, ulp_e);
    float q = rintf(a / ulp) * ulp;
    if (q == 0.0f) return sgn;
    unsigned qu = __float_as_uint(q);
    int qe = (int)((qu >> 23) & 0xFF) - 127;
    unsigned code;
    if (qe < -6) code = (unsigned)(q * 512.0f + 0.5f);
    else {
        code = ((unsigned)(qe + 7) << 3) | ((qu >> 20) & 7);
        if (code > 0x7Eu) code = 0x7Eu;
    }
    return sgn | code;
}

static __device__ __forceinline__ unsigned pk_fp8_lo(float a, float b, unsigned old) {
#if __has_builtin(__builtin_amdgcn_cvt_pk_fp8_f32)
    return (unsigned)__builtin_amdgcn_cvt_pk_fp8_f32(a, b, (int)old, false);
#else
    unsigned p = f32_to_e4m3_sw(a) | (f32_to_e4m3_sw(b) << 8);
    return (old & 0xFFFF0000u) | p;
#endif
}

static __device__ __forceinline__ unsigned pk_fp8_hi(float a, float b, unsigned old) {
#if __has_builtin(__builtin_amdgcn_cvt_pk_fp8_f32)
    return (unsigned)__builtin_amdgcn_cvt_pk_fp8_f32(a, b, (int)old, true);
#else
    unsigned p = f32_to_e4m3_sw(a) | (f32_to_e4m3_sw(b) << 8);
    return (old & 0x0000FFFFu) | (p << 16);
#endif
}

static __device__ __forceinline__ float bf16_round(float f) {
    unsigned u = __float_as_uint(f);
    unsigned r = (u + 0x7FFFu + ((u >> 16) & 1u)) & 0xFFFF0000u;
    return __uint_as_float(r);
}

__global__ __launch_bounds__(256)
void quant_x_kernel(const float* __restrict__ x, uint8_t* __restrict__ fragA,
                    float* __restrict__ sxT, int M) {
    const int bid  = blockIdx.x;
    const int mt16 = bid / KT;
    const int kt   = bid % KT;
    const int t = threadIdx.x;
    const int r = t >> 4;
    const int c = t & 15;
    const int m = mt16 * 16 + r;

    const float4 v = *(const float4*)(x + (size_t)m * KDIM + kt * 64 + c * 4);
    float amax = fmaxf(fmaxf(fabsf(v.x), fabsf(v.y)), fmaxf(fabsf(v.z), fabsf(v.w)));
    #pragma unroll
    for (int off = 1; off < 16; off <<= 1)
        amax = fmaxf(amax, __shfl_xor(amax, off, 32));

    const float am    = fmaxf(amax, 1e-12f);
    const float scale = am * (1.0f / 448.0f);
    const float inv   = 448.0f / am;

    unsigned d = pk_fp8_lo(v.x * inv, v.y * inv, 0u);
    d          = pk_fp8_hi(v.z * inv, v.w * inv, d);

    const int vq = ((c >> 3) << 2) | (((c >> 2) & 1) << 1) | (c & 1);
    const int h  = (c >> 1) & 1;
    unsigned* frag = (unsigned*)(fragA + ((size_t)(mt16 * KT + kt)) * 1024);
    frag[(h * 16 + r) * 8 + vq] = d;

    if (c == 0) sxT[(size_t)kt * M + m] = scale;
}

__global__ __launch_bounds__(256)
void quant_w_kernel(const float* __restrict__ w, uint8_t* __restrict__ fragB,
                    float* __restrict__ swT) {
    __shared__ float red[256];
    const int n64 = blockIdx.x;
    const int kt  = blockIdx.y;
    const int t = threadIdx.x;
    const int rl = t >> 2;
    const int cq = t & 3;
    const int n = n64 * 64 + rl;
    const float* pw = w + (size_t)n * KDIM + kt * 64;

    float4 vv[4];
    float amax = 0.0f;
    #pragma unroll
    for (int j = 0; j < 4; ++j) {
        vv[j] = *(const float4*)(pw + (cq + 4 * j) * 4);
        amax = fmaxf(amax, fmaxf(fmaxf(fabsf(vv[j].x), fabsf(vv[j].y)),
                                 fmaxf(fabsf(vv[j].z), fabsf(vv[j].w))));
    }
    red[t] = amax;
    __syncthreads();
    #pragma unroll
    for (int s2 = 128; s2 > 0; s2 >>= 1) {
        if (t < s2) red[t] = fmaxf(red[t], red[t + s2]);
        __syncthreads();
    }
    const float am    = fmaxf(red[0], 1e-12f);
    const float scale = am * (1.0f / 448.0f);
    const float inv   = 448.0f / am;

    const int nt16 = n64 * 4 + (rl >> 4);
    const int nn   = rl & 15;
    unsigned* frag = (unsigned*)(fragB + ((size_t)(nt16 * KT + kt)) * 1024);
    #pragma unroll
    for (int j = 0; j < 4; ++j) {
        const int c = cq + 4 * j;
        unsigned d = pk_fp8_lo(vv[j].x * inv, vv[j].y * inv, 0u);
        d          = pk_fp8_hi(vv[j].z * inv, vv[j].w * inv, d);
        const int vq = ((c >> 3) << 2) | (((c >> 2) & 1) << 1) | (c & 1);
        const int h  = (c >> 1) & 1;
        frag[(h * 16 + nn) * 8 + vq] = d;
    }
    if (t == 0) swT[(size_t)kt * NB64 + n64] = scale;
}

__global__ __launch_bounds__(256)
void fp8_gemm_kernel(const uint8_t* __restrict__ fragA,
                     const uint8_t* __restrict__ fragB,
                     const float* __restrict__ sxT,
                     const float* __restrict__ swT,
                     const float* __restrict__ bias,
                     float* __restrict__ out, int M) {
    const int lane = threadIdx.x & 31;
    const int wave = threadIdx.x >> 5;
    const int wm = wave & 3;
    const int wn = wave >> 2;
    const int m0 = blockIdx.y * 128 + wm * 32;
    const int n0 = blockIdx.x * 128 + wn * 64;
    const int mt0 = m0 >> 4;
    const int nt0 = n0 >> 4;

    const int* pA0 = (const int*)fragA + (size_t)(mt0 + 0) * KT * 256 + lane * 8;
    const int* pA1 = (const int*)fragA + (size_t)(mt0 + 1) * KT * 256 + lane * 8;
    const int* pB0 = (const int*)fragB + (size_t)(nt0 + 0) * KT * 256 + lane * 8;
    const int* pB1 = (const int*)fragB + (size_t)(nt0 + 1) * KT * 256 + lane * 8;
    const int* pB2 = (const int*)fragB + (size_t)(nt0 + 2) * KT * 256 + lane * 8;
    const int* pB3 = (const int*)fragB + (size_t)(nt0 + 3) * KT * 256 + lane * 8;

    v8f_t acc[2][4];
    #pragma unroll
    for (int a = 0; a < 2; ++a)
        #pragma unroll
        for (int b = 0; b < 4; ++b)
            #pragma unroll
            for (int i = 0; i < 8; ++i) acc[a][b][i] = 0.0f;

    v8f_t zc;
    #pragma unroll
    for (int i = 0; i < 8; ++i) zc[i] = 0.0f;

    const int hsel  = lane >> 4;
    const int mrow0 = m0 + hsel * 8;
    const int nb64  = n0 >> 6;

    struct Frags  { v8i_t a0, a1, b0, b1, b2, b3; };
    struct Scales { float4 xa, xb, xc, xd; float w; };

    auto loadFrags = [&](int kt, Frags& f) {
        const size_t off = (size_t)kt * 256;
        f.a0 = *(const v8i_t*)(pA0 + off);
        f.a1 = *(const v8i_t*)(pA1 + off);
        f.b0 = *(const v8i_t*)(pB0 + off);
        f.b1 = *(const v8i_t*)(pB1 + off);
        f.b2 = *(const v8i_t*)(pB2 + off);
        f.b3 = *(const v8i_t*)(pB3 + off);
        __builtin_prefetch((const void*)(pA0 + off + 8 * 256), 0, 1);
        __builtin_prefetch((const void*)(pB0 + off + 8 * 256), 0, 1);
        __builtin_prefetch((const void*)(pB2 + off + 8 * 256), 0, 1);
    };

    auto loadScales = [&](int kt, Scales& s) {
        const float* sx = sxT + (size_t)kt * M;
        s.xa = *(const float4*)(sx + mrow0);
        s.xb = *(const float4*)(sx + mrow0 + 4);
        s.xc = *(const float4*)(sx + mrow0 + 16);
        s.xd = *(const float4*)(sx + mrow0 + 20);
        s.w  = swT[(size_t)kt * NB64 + nb64];
    };

    auto compute = [&](const Frags& f, const Scales& s) {
        float c0[8] = {s.xa.x * s.w, s.xa.y * s.w, s.xa.z * s.w, s.xa.w * s.w,
                       s.xb.x * s.w, s.xb.y * s.w, s.xb.z * s.w, s.xb.w * s.w};
        float c1[8] = {s.xc.x * s.w, s.xc.y * s.w, s.xc.z * s.w, s.xc.w * s.w,
                       s.xd.x * s.w, s.xd.y * s.w, s.xd.z * s.w, s.xd.w * s.w};
        v8i_t bv[4] = {f.b0, f.b1, f.b2, f.b3};
        #pragma unroll
        for (int nt = 0; nt < 4; ++nt) {
            v8f_t p0 = __builtin_amdgcn_wmma_f32_16x16x64_fp8_fp8(
                f.a0, bv[nt], (short)0, zc, false, false);
            v8f_t p1 = __builtin_amdgcn_wmma_f32_16x16x64_fp8_fp8(
                f.a1, bv[nt], (short)0, zc, false, false);
            #pragma unroll
            for (int i = 0; i < 8; ++i) {
                acc[0][nt][i] = fmaf(p0[i], c0[i], acc[0][nt][i]);
                acc[1][nt][i] = fmaf(p1[i], c1[i], acc[1][nt][i]);
            }
        }
    };

    Frags fX, fY;
    Scales sX, sY;
    loadFrags(0, fX);
    loadScales(0, sX);

    int kt = 0;
    #pragma nounroll
    for (; kt + 2 <= KT - 1; kt += 2) {
        loadFrags(kt + 1, fY);
        loadScales(kt + 1, sY);
        compute(fX, sX);
        loadFrags(kt + 2, fX);
        loadScales(kt + 2, sX);
        compute(fY, sY);
    }
    loadFrags(KT - 1, fY);
    loadScales(KT - 1, sY);
    compute(fX, sX);
    compute(fY, sY);

    const int colb = lane & 15;
    #pragma unroll
    for (int mt = 0; mt < 2; ++mt) {
        const int rbase = m0 + mt * 16 + hsel * 8;
        #pragma unroll
        for (int nt = 0; nt < 4; ++nt) {
            const int col = n0 + nt * 16 + colb;
            const float bvs = bias[col];
            #pragma unroll
            for (int i = 0; i < 8; ++i) {
                out[(size_t)(rbase + i) * NDIM + col] =
                    bf16_round(acc[mt][nt][i]) + bvs;
            }
        }
    }
}

extern "C" void kernel_launch(void* const* d_in, const int* in_sizes, int n_in,
                              void* d_out, int out_size, void* d_ws, size_t ws_size,
                              hipStream_t stream) {
    const float* x    = (const float*)d_in[0];
    const float* w    = (const float*)d_in[1];
    const float* bias = (const float*)d_in[2];
    float* out = (float*)d_out;
    const int M = in_sizes[0] / KDIM;

    uint8_t* ws    = (uint8_t*)d_ws;
    uint8_t* fragA = ws;
    uint8_t* fragB = fragA + (size_t)M * KDIM;
    float*   sxT   = (float*)(fragB + (size_t)NDIM * KDIM);
    float*   swT   = sxT + (size_t)KT * M;

    quant_x_kernel<<<(M / 16) * KT, 256, 0, stream>>>(x, fragA, sxT, M);
    quant_w_kernel<<<dim3(NB64, KT), 256, 0, stream>>>(w, fragB, swT);
    fp8_gemm_kernel<<<dim3(NDIM / 128, M / 128), 256, 0, stream>>>(
        fragA, fragB, sxT, swT, bias, out, M);
}
